// Net_66279935312060
// MI455X (gfx1250) — matrix operand order repaired and checked by reading, not compiled, not run on hardware
//
#include <hip/hip_runtime.h>
#include <hip/hip_bf16.h>

typedef __attribute__((ext_vector_type(16))) _Float16 v16h;
typedef __attribute__((ext_vector_type(8)))  float    v8f;

#define N_PFC 8192
#define N_VTX 2048
#define G     32
#define H     32
#define K_NN  16

union Frag { v16h h; unsigned int u[8]; };

__device__ __forceinline__ float lrelu(float x) { return fmaxf(x, 0.01f * x); }

// ---------------------------------------------------------------------------
// Group boundaries: start[g] = lower_bound(batch, g) for g in [0, G]
// ---------------------------------------------------------------------------
__global__ void bounds_kernel(const int* __restrict__ batch, int n,
                              int* __restrict__ start) {
    int g = blockIdx.x * blockDim.x + threadIdx.x;
    if (g > G) return;
    int lo = 0, hi = n;
    while (lo < hi) {
        int mid = (lo + hi) >> 1;
        if (batch[mid] < g) lo = mid + 1; else hi = mid;
    }
    start[g] = lo;
}

// ---------------------------------------------------------------------------
// Encoder MLP (kept f32 VALU: its output feeds the f32 kNN distance ranking)
// ---------------------------------------------------------------------------
template <int IN>
__global__ void encode_kernel(const float* __restrict__ x, int n,
                              const float* __restrict__ w1, const float* __restrict__ b1,
                              const float* __restrict__ w2, const float* __restrict__ b2,
                              float* __restrict__ out) {
    int i = blockIdx.x * blockDim.x + threadIdx.x;
    if (i >= n) return;
    float xi[IN];
#pragma unroll
    for (int r = 0; r < IN; ++r) xi[r] = x[i * IN + r];
    float h1[H];
#pragma unroll
    for (int c = 0; c < H; ++c) {
        float a = b1[c];
#pragma unroll
        for (int r = 0; r < IN; ++r) a = fmaf(xi[r], w1[r * H + c], a);
        h1[c] = lrelu(a);
    }
#pragma unroll
    for (int c = 0; c < H; ++c) {
        float a = b2[c];
#pragma unroll
        for (int r = 0; r < H; ++r) a = fmaf(h1[r], w2[r * H + c], a);
        out[i * H + c] = lrelu(a);
    }
}

// ---------------------------------------------------------------------------
// kNN (k=16) within batch group; sorted batches -> contiguous candidate span.
// Rank by |s|^2 - 2 d.s (same ordering as full d^2). Register insertion sort.
// ---------------------------------------------------------------------------
__global__ void knn_kernel(const float* __restrict__ dstX,
                           const int*   __restrict__ batchDst, int nDst,
                           const float* __restrict__ srcX,
                           const int*   __restrict__ srcStart,  // [G+1]
                           int* __restrict__ idxOut) {
    int i = blockIdx.x * blockDim.x + threadIdx.x;
    if (i >= nDst) return;
    int g = batchDst[i];
    int s = srcStart[g], e = srcStart[g + 1];

    float t[H];  // -2 * dst feature
    const float4* dp = (const float4*)(dstX + (size_t)i * H);
#pragma unroll
    for (int q = 0; q < H / 4; ++q) {
        float4 v = dp[q];
        t[4 * q + 0] = -2.0f * v.x; t[4 * q + 1] = -2.0f * v.y;
        t[4 * q + 2] = -2.0f * v.z; t[4 * q + 3] = -2.0f * v.w;
    }

    float bestD[K_NN];
    int   bestI[K_NN];
#pragma unroll
    for (int p = 0; p < K_NN; ++p) { bestD[p] = INFINITY; bestI[p] = s; }

    for (int j = s; j < e; ++j) {
        const float4* sp = (const float4*)(srcX + (size_t)j * H);
        float sc = 0.0f;
#pragma unroll
        for (int q = 0; q < H / 4; ++q) {
            float4 v = sp[q];
            sc = fmaf(v.x, v.x + t[4 * q + 0], sc);
            sc = fmaf(v.y, v.y + t[4 * q + 1], sc);
            sc = fmaf(v.z, v.z + t[4 * q + 2], sc);
            sc = fmaf(v.w, v.w + t[4 * q + 3], sc);
        }
        float v = sc; int id = j;
#pragma unroll
        for (int p = 0; p < K_NN; ++p) {
            if (v < bestD[p]) {
                float td = bestD[p]; int ti = bestI[p];
                bestD[p] = v; bestI[p] = id; v = td; id = ti;
            }
        }
    }
#pragma unroll
    for (int p = 0; p < K_NN; ++p) idxOut[(size_t)i * K_NN + p] = bestI[p];
}

// ---------------------------------------------------------------------------
// Fused EdgeConv: one wave per dst point. 16 edges x 64 feats tile in LDS
// (f16), conv_w transposed in LDS (f16). 2 k-steps x 2 n-tiles of
// v_wmma_f32_16x16x32_f16, then bias + lrelu + max over 16 neighbors.
// ---------------------------------------------------------------------------
__global__ __launch_bounds__(256) void edgeconv_kernel(
    const float* __restrict__ dstX,   // [nDst][32]
    const float* __restrict__ srcX,   // [nSrc][32]
    const int*   __restrict__ idx,    // [nDst][16]
    const float* __restrict__ convW,  // [64][32]
    const float* __restrict__ convB,  // [32]
    float* __restrict__ outF) {       // [nDst][32]
    __shared__ _Float16 wT[32 * 64];        // wT[n*64 + k]
    __shared__ _Float16 eT[8][16 * 64];     // per-wave edge tile [row*64 + col]
    __shared__ float    bS[32];

    int tid = threadIdx.x;
    for (int t = tid; t < 64 * 32; t += 256) {
        int k = t >> 5, n = t & 31;
        wT[n * 64 + k] = (_Float16)convW[t];
    }
    if (tid < 32) bS[tid] = convB[tid];

    int wv = tid >> 5, lane = tid & 31;
    int r = lane & 15, half = lane >> 4;
    int i = blockIdx.x * 8 + wv;           // nDst == gridDim.x*8 exactly

    const float* xi = dstX + (size_t)i * H;
    _Float16* e = eT[wv];
    if (half == 0) {
#pragma unroll
        for (int c = 0; c < H; ++c) e[r * 64 + c] = (_Float16)xi[c];
    } else {
        int j = idx[(size_t)i * K_NN + r];
        const float* xj = srcX + (size_t)j * H;
#pragma unroll
        for (int c = 0; c < H; ++c) e[r * 64 + 32 + c] = (_Float16)(xj[c] - xi[c]);
    }
    __syncthreads();

    const unsigned int* eu = (const unsigned int*)e;   // f16 pair per u32
    const unsigned int* wu = (const unsigned int*)wT;

    v8f acc0 = {}; v8f acc1 = {};
#pragma unroll
    for (int kb = 0; kb < 2; ++kb) {
        Frag a;
        int aBase = r * 32 + half * 8 + kb * 16;
#pragma unroll
        for (int p = 0; p < 4; ++p) a.u[p] = eu[aBase + p];
#pragma unroll
        for (int p = 0; p < 4; ++p) a.u[4 + p] = eu[aBase + 4 + p];
        Frag b0, b1;
        int kBase = kb * 16 + half * 8;
#pragma unroll
        for (int p = 0; p < 8; ++p) {
            b0.u[p] = wu[r * 32 + kBase + p];
            b1.u[p] = wu[(16 + r) * 32 + kBase + p];
        }
        acc0 = __builtin_amdgcn_wmma_f32_16x16x32_f16(false, a.h, false, b0.h,
                                                      (short)0, acc0, false, false);
        acc1 = __builtin_amdgcn_wmma_f32_16x16x32_f16(false, a.h, false, b1.h,
                                                      (short)0, acc1, false, false);
    }

    float bv0 = bS[r], bv1 = bS[16 + r];
    float m0 = -INFINITY, m1 = -INFINITY;
#pragma unroll
    for (int p = 0; p < 8; ++p) {
        m0 = fmaxf(m0, lrelu(acc0[p] + bv0));
        m1 = fmaxf(m1, lrelu(acc1[p] + bv1));
    }
    m0 = fmaxf(m0, __shfl_xor(m0, 16, 32));
    m1 = fmaxf(m1, __shfl_xor(m1, 16, 32));
    if (half == 0) {
        outF[(size_t)i * H + r]      = m0;
        outF[(size_t)i * H + 16 + r] = m1;
    }
}

// ---------------------------------------------------------------------------
// Head MLP on the WMMA path: one wave per 16 rows.
//   layer1: 32 -> 64  = 1 k-step x 4 n-tiles  (4 WMMAs)
//   layer2: 64 -> 32  = 2 k-steps x 2 n-tiles (4 WMMAs)
//   layers 3/4 (32->4->1): tiny, VALU on lanes 0-15.
// h1/h2 staged as f16 LDS tiles between layers; biases+lrelu fused.
// ---------------------------------------------------------------------------
__global__ __launch_bounds__(256) void head_kernel(
    const float* __restrict__ feats,  // [n][32]
    const float* __restrict__ w1, const float* __restrict__ b1,  // (32,64)
    const float* __restrict__ w2, const float* __restrict__ b2,  // (64,32)
    const float* __restrict__ w3, const float* __restrict__ b3,  // (32,4)
    const float* __restrict__ w4, const float* __restrict__ b4,  // (4,1)
    const int* __restrict__ batch,
    float* __restrict__ out, int n) {
    __shared__ _Float16 w1T[64 * 32];       // w1T[nn*32 + k]
    __shared__ _Float16 w2T[32 * 64];       // w2T[nn*64 + k]
    __shared__ float    b1S[64], b2S[32];
    __shared__ _Float16 tA[8][16 * 32];     // in0, then h2
    __shared__ _Float16 tB[8][16 * 64];     // h1

    int tid = threadIdx.x;
    for (int t = tid; t < 32 * 64; t += 256) {
        int k = t >> 6, nn = t & 63;
        w1T[nn * 32 + k] = (_Float16)w1[t];
    }
    for (int t = tid; t < 64 * 32; t += 256) {
        int k = t >> 5, nn = t & 31;
        w2T[nn * 64 + k] = (_Float16)w2[t];
    }
    if (tid < 64) b1S[tid] = b1[tid];
    if (tid < 32) b2S[tid] = b2[tid];

    int wv = tid >> 5, lane = tid & 31;
    int r = lane & 15, half = lane >> 4;
    int i0 = blockIdx.x * 128 + wv * 16;    // n == gridDim.x*128 exactly

    // stage input tile (16 x 32) as f16
    _Float16* ta = tA[wv];
    _Float16* tb = tB[wv];
#pragma unroll
    for (int c = 0; c < 16; ++c) {
        int col = half * 16 + c;
        ta[r * 32 + col] = (_Float16)feats[(size_t)(i0 + r) * 32 + col];
    }
    __syncthreads();

    // ----- layer 1: h1 = lrelu(in0 @ w1 + b1), 16x32 @ 32x64 -----
    {
        const unsigned int* au  = (const unsigned int*)ta;   // row stride 16 u32
        const unsigned int* w1u = (const unsigned int*)w1T;  // n stride 16 u32
        Frag a;
        int aBase = r * 16 + half * 8;
#pragma unroll
        for (int p = 0; p < 4; ++p) a.u[p] = au[aBase + p];
#pragma unroll
        for (int p = 0; p < 4; ++p) a.u[4 + p] = au[aBase + 4 + p];
#pragma unroll
        for (int nt = 0; nt < 4; ++nt) {
            int n0 = nt * 16;
            Frag b;
#pragma unroll
            for (int p = 0; p < 8; ++p) b.u[p] = w1u[(n0 + r) * 16 + half * 8 + p];
            v8f acc = {};
            acc = __builtin_amdgcn_wmma_f32_16x16x32_f16(false, a.h, false, b.h,
                                                         (short)0, acc, false, false);
            float bv = b1S[n0 + r];
#pragma unroll
            for (int p = 0; p < 8; ++p)
                tb[(p + half * 8) * 64 + n0 + r] = (_Float16)lrelu(acc[p] + bv);
        }
    }
    __syncthreads();

    // ----- layer 2: h2 = lrelu(h1 @ w2 + b2), 16x64 @ 64x32 -----
    {
        const unsigned int* hu  = (const unsigned int*)tb;   // row stride 32 u32
        const unsigned int* w2u = (const unsigned int*)w2T;  // n stride 32 u32
        v8f acc2[2] = {{}, {}};
#pragma unroll
        for (int kb = 0; kb < 2; ++kb) {
            Frag a;
            int aBase = r * 32 + half * 8 + kb * 16;
#pragma unroll
            for (int p = 0; p < 4; ++p) a.u[p] = hu[aBase + p];
#pragma unroll
            for (int p = 0; p < 4; ++p) a.u[4 + p] = hu[aBase + 4 + p];
#pragma unroll
            for (int nt = 0; nt < 2; ++nt) {
                Frag b;
#pragma unroll
                for (int p = 0; p < 8; ++p)
                    b.u[p] = w2u[(nt * 16 + r) * 32 + kb * 16 + half * 8 + p];
                acc2[nt] = __builtin_amdgcn_wmma_f32_16x16x32_f16(
                    false, a.h, false, b.h, (short)0, acc2[nt], false, false);
            }
        }
#pragma unroll
        for (int nt = 0; nt < 2; ++nt) {
            float bv = b2S[nt * 16 + r];
#pragma unroll
            for (int p = 0; p < 8; ++p)
                ta[(p + half * 8) * 32 + nt * 16 + r] = (_Float16)lrelu(acc2[nt][p] + bv);
        }
    }
    __syncthreads();

    // ----- layers 3/4: 32 -> 4 -> 1, lanes 0-15, f32 VALU -----
    if (half == 0) {
        int ig = i0 + r;
        float h2v[32];
#pragma unroll
        for (int c = 0; c < 32; ++c) h2v[c] = (float)ta[r * 32 + c];
        float h3[4];
#pragma unroll
        for (int c = 0; c < 4; ++c) {
            float a = b3[c];
#pragma unroll
            for (int k = 0; k < 32; ++k) a = fmaf(h2v[k], w3[k * 4 + c], a);
            h3[c] = lrelu(a);
        }
        float o = b4[0];
#pragma unroll
        for (int c = 0; c < 4; ++c) o = fmaf(h3[c], w4[c], o);
        out[ig] = lrelu(o);
        ((int*)out)[n + ig] = batch[ig];   // second tuple element, bit-preserving
    }
}

// ---------------------------------------------------------------------------
extern "C" void kernel_launch(void* const* d_in, const int* in_sizes, int n_in,
                              void* d_out, int out_size, void* d_ws, size_t ws_size,
                              hipStream_t stream) {
    const float* x_pfc     = (const float*)d_in[0];
    const float* x_vtx     = (const float*)d_in[1];
    const int*   batch_pfc = (const int*)d_in[2];
    const int*   batch_vtx = (const int*)d_in[3];
    const float* pfc_w1 = (const float*)d_in[4];
    const float* pfc_b1 = (const float*)d_in[5];
    const float* pfc_w2 = (const float*)d_in[6];
    const float* pfc_b2 = (const float*)d_in[7];
    const float* vtx_w1 = (const float*)d_in[8];
    const float* vtx_b1 = (const float*)d_in[9];
    const float* vtx_w2 = (const float*)d_in[10];
    const float* vtx_b2 = (const float*)d_in[11];
    const float* conv_w = (const float*)d_in[12];
    const float* conv_b = (const float*)d_in[13];
    const float* out_w1 = (const float*)d_in[14];
    const float* out_b1 = (const float*)d_in[15];
    const float* out_w2 = (const float*)d_in[16];
    const float* out_b2 = (const float*)d_in[17];
    const float* out_w3 = (const float*)d_in[18];
    const float* out_b3 = (const float*)d_in[19];
    const float* out_w4 = (const float*)d_in[20];
    const float* out_b4 = (const float*)d_in[21];

    char* base = (char*)d_ws;
    size_t off = 0;
    auto carve = [&](size_t bytes) {
        void* p = base + off;
        off = (off + bytes + 255) & ~(size_t)255;
        return p;
    };
    float* pfcEnc  = (float*)carve((size_t)N_PFC * H * 4);
    float* vtxEnc  = (float*)carve((size_t)N_VTX * H * 4);
    float* feats1  = (float*)carve((size_t)N_PFC * H * 4);
    float* feats2  = (float*)carve((size_t)N_PFC * H * 4);
    int*   idx1    = (int*)  carve((size_t)N_PFC * K_NN * 4);
    int*   idx2    = (int*)  carve((size_t)N_PFC * K_NN * 4);
    int*   pfcBnd  = (int*)  carve((G + 1) * 4);
    int*   vtxBnd  = (int*)  carve((G + 1) * 4);

    bounds_kernel<<<1, 64, 0, stream>>>(batch_pfc, N_PFC, pfcBnd);
    bounds_kernel<<<1, 64, 0, stream>>>(batch_vtx, N_VTX, vtxBnd);

    encode_kernel<7><<<N_PFC / 256, 256, 0, stream>>>(x_pfc, N_PFC,
        pfc_w1, pfc_b1, pfc_w2, pfc_b2, pfcEnc);
    encode_kernel<4><<<N_VTX / 256, 256, 0, stream>>>(x_vtx, N_VTX,
        vtx_w1, vtx_b1, vtx_w2, vtx_b2, vtxEnc);

    knn_kernel<<<N_PFC / 256, 256, 0, stream>>>(pfcEnc, batch_pfc, N_PFC,
                                                pfcEnc, pfcBnd, idx1);
    edgeconv_kernel<<<N_PFC / 8, 256, 0, stream>>>(pfcEnc, pfcEnc, idx1,
                                                   conv_w, conv_b, feats1);

    knn_kernel<<<N_PFC / 256, 256, 0, stream>>>(feats1, batch_pfc, N_PFC,
                                                vtxEnc, vtxBnd, idx2);
    edgeconv_kernel<<<N_PFC / 8, 256, 0, stream>>>(feats1, vtxEnc, idx2,
                                                   conv_w, conv_b, feats2);

    head_kernel<<<N_PFC / 128, 256, 0, stream>>>(feats2,
        out_w1, out_b1, out_w2, out_b2, out_w3, out_b3, out_w4, out_b4,
        batch_pfc, (float*)d_out, N_PFC);
}
